// OnlineSlidingWindowAttn_55147380081134
// MI455X (gfx1250) — matrix operand order repaired and checked by reading, not compiled, not run on hardware
//
#include <hip/hip_runtime.h>

typedef __attribute__((ext_vector_type(16))) _Float16 v16h;
typedef __attribute__((ext_vector_type(8)))  _Float16 v8h;
typedef __attribute__((ext_vector_type(8)))  float    v8f;

#define HEAD_DIM    128
#define NUM_Q_HEAD  32
#define NUM_KV_HEAD 8
#define SEQLEN_Q    4096
#define SEQLEN_KV   4096
#define BQ          128
#define BKV         128
#define WINDOW      512
#define SOFT_CAP    50.0f
#define EPS_RMS     1e-5f
#define QK_SCALE    0.08838834764831845f   /* 1/sqrt(128) */

// LDS layout (aliased regions, 132608 B total):
//   [0      , 32768) : qh f16[128][128]   (phase 0-1)  --> vh f16[128][128] (phase 2-3)
//   [32768  , 65536) : kh f16[128][128]   (phase 0-1)  --> ph f16[128][128] (phase 2-3)
//   [65536  ,131072) : S  f32[128][128]   (raw logits, then capped/masked logits)
//   [131072 ,132608) : lseb/cold/cblk f32[128] each
#define SMEM_BYTES (2*BQ*HEAD_DIM*2 + BQ*BKV*4 + 3*BQ*4)

__global__ __launch_bounds__(256) void oswa_block_kernel(
    const float* __restrict__ q,  const float* __restrict__ kk, const float* __restrict__ v,
    const float* __restrict__ g_o, const float* __restrict__ g_lse,
    const float* __restrict__ gq, const float* __restrict__ gk,
    const int* __restrict__ p_biq, const int* __restrict__ p_bikv,
    float* __restrict__ out_o, float* __restrict__ out_lse)
{
    __shared__ alignas(16) unsigned char smem[SMEM_BYTES];
    _Float16 (*qh)[HEAD_DIM] = (_Float16 (*)[HEAD_DIM])(smem);
    _Float16 (*vh)[HEAD_DIM] = (_Float16 (*)[HEAD_DIM])(smem);            // alias of qh
    _Float16 (*kh)[HEAD_DIM] = (_Float16 (*)[HEAD_DIM])(smem + 32768);
    _Float16 (*ph)[BKV]      = (_Float16 (*)[BKV])(smem + 32768);         // alias of kh
    float    (*S)[BKV]       = (float    (*)[BKV])(smem + 65536);
    float*   lseb            = (float*)(smem + 131072);
    float*   cold            = lseb + BQ;
    float*   cblk            = cold + BQ;

    const float INF = __builtin_inff();

    const int bh  = blockIdx.x;
    const int b   = bh / NUM_Q_HEAD;
    const int h   = bh % NUM_Q_HEAD;
    const int hk  = h / (NUM_Q_HEAD / NUM_KV_HEAD);
    const int tid = threadIdx.x;
    const int lane = tid & 31;
    const int wv   = tid >> 5;        // wave 0..7
    const int ln   = lane & 15;
    const int g    = lane >> 4;       // half-wave select

    const int biq  = p_biq[0];
    const int bikv = p_bikv[0];
    const int row0 = biq * BQ;        // unclamped (used for mask positions)
    const int col0 = bikv * BKV;
    int row0m = row0;                 // clamped slice base (dynamic_slice semantics)
    if (row0m < 0) row0m = 0;
    if (row0m > SEQLEN_Q - BQ) row0m = SEQLEN_Q - BQ;

    // ---------------- Phase 0: fused GroupRMSNorm of Q and K -> f16 LDS ----------
    if (tid < BQ) {
        const int i = tid;
        const float* qr = q + (((size_t)b * BQ + i) * NUM_Q_HEAD + h) * HEAD_DIM;
        float ss = 0.f;
        for (int d = 0; d < HEAD_DIM; ++d) { float x = qr[d]; ss += x * x; }
        const float inv = rsqrtf(ss * (1.0f / HEAD_DIM) + EPS_RMS);
        for (int d = 0; d < HEAD_DIM; ++d)
            qh[i][d] = (_Float16)(qr[d] * inv * gq[h * HEAD_DIM + d]);
    } else {
        const int j = tid - BQ;
        const float* kr = kk + (((size_t)b * BKV + j) * NUM_KV_HEAD + hk) * HEAD_DIM;
        float ss = 0.f;
        for (int d = 0; d < HEAD_DIM; ++d) { float x = kr[d]; ss += x * x; }
        const float inv = rsqrtf(ss * (1.0f / HEAD_DIM) + EPS_RMS);
        for (int d = 0; d < HEAD_DIM; ++d)
            kh[j][d] = (_Float16)(kr[d] * inv * gk[hk * HEAD_DIM + d]);
    }
    __syncthreads();

    // ---------------- Phase 1: S = qn @ kn^T via v_wmma_f32_16x16x32_f16 ---------
    {
        const int m0 = wv * 16;
        const _Float16* arow = &qh[m0 + ln][0];
        for (int nt = 0; nt < 8; ++nt) {
            const int n0 = nt * 16;
            const _Float16* brow = &kh[n0 + ln][0];
            v8f acc = {};
#pragma unroll
            for (int kt = 0; kt < 4; ++kt) {
                const int k0 = kt * 32;
                v16h a = *(const v16h*)(arow + k0 + 16 * g);
                v16h bb = *(const v16h*)(brow + k0 + 16 * g);
                acc = __builtin_amdgcn_wmma_f32_16x16x32_f16(
                    false, a, false, bb, (short)0, acc, false, false);
            }
#pragma unroll
            for (int r = 0; r < 8; ++r)
                S[m0 + r + 8 * g][n0 + ln] = acc[r];
        }
    }
    __syncthreads();

    // ---------------- Phase 2: V -> f16 LDS (reuses qh); cap/mask/softmax/merge stats
    for (int e = tid; e < BKV * HEAD_DIM; e += 256) {
        const int j = e >> 7, d = e & 127;
        vh[j][d] = (_Float16)(v[(((size_t)b * BKV + j) * NUM_KV_HEAD + hk) * HEAD_DIM + d]);
    }
    if (tid < BQ) {
        const int i  = tid;
        const int qi = row0 + i;
        const int diag = qi + (SEQLEN_KV - SEQLEN_Q);
        float mx = -INF;
        for (int j = 0; j < BKV; ++j) {
            const int kj = col0 + j;
            const bool ok = (kj <= diag) && (kj >= diag - WINDOW) &&
                            (qi < SEQLEN_Q) && (kj < SEQLEN_KV);
            const float l = ok ? SOFT_CAP * tanhf(S[i][j] * (QK_SCALE / SOFT_CAP)) : -INF;
            S[i][j] = l;
            mx = fmaxf(mx, l);
        }
        float lse;
        if (mx > -INF) {
            float sum = 0.f;
            for (int j = 0; j < BKV; ++j) sum += expf(S[i][j] - mx);
            lse = mx + logf(sum);
            for (int j = 0; j < BKV; ++j) ph[i][j] = (_Float16)expf(S[i][j] - lse);
        } else {
            lse = -INF;
            for (int j = 0; j < BKV; ++j) ph[i][j] = (_Float16)0.0f;
        }
        lseb[i] = lse;
        // online merge statistics (logaddexp with -inf handling)
        const size_t li = ((size_t)b * NUM_Q_HEAD + h) * SEQLEN_Q + (size_t)(row0m + i);
        const float lo = g_lse[li];
        float lnew;
        if (lo == -INF)       lnew = lse;
        else if (lse == -INF) lnew = lo;
        else { const float mm = fmaxf(lo, lse);
               lnew = mm + logf(expf(lo - mm) + expf(lse - mm)); }
        const bool fin = (lnew == lnew) && (lnew > -INF) && (lnew < INF);
        cold[i] = fin ? expf(lo  - lnew) : 0.0f;
        cblk[i] = fin ? expf(lse - lnew) : 0.0f;
        out_lse[li] = lnew;
    }
    __syncthreads();

    // ---------------- Phase 3: O = P @ V via WMMA, fused merge + writeback -------
    {
        const int m0 = wv * 16;
        const _Float16* arow = &ph[m0 + ln][0];
        for (int nt = 0; nt < 8; ++nt) {
            const int n0 = nt * 16;
            v8f acc = {};
#pragma unroll
            for (int kt = 0; kt < 4; ++kt) {
                const int k0 = kt * 32;
                v16h a = *(const v16h*)(arow + k0 + 16 * g);
                v16h bb;
#pragma unroll
                for (int s = 0; s < 16; ++s) bb[s] = vh[k0 + s + 16 * g][n0 + ln];
                acc = __builtin_amdgcn_wmma_f32_16x16x32_f16(
                    false, a, false, bb, (short)0, acc, false, false);
            }
#pragma unroll
            for (int r = 0; r < 8; ++r) {
                const int i    = m0 + r + 8 * g;
                const int dcol = n0 + ln;
                const int qrow = row0m + i;
                const size_t oidx =
                    (((size_t)b * SEQLEN_Q + qrow) * NUM_Q_HEAD + h) * HEAD_DIM + dcol;
                out_o[oidx] = cold[i] * g_o[oidx] + cblk[i] * acc[r];
            }
        }
    }
}

extern "C" void kernel_launch(void* const* d_in, const int* in_sizes, int n_in,
                              void* d_out, int out_size, void* d_ws, size_t ws_size,
                              hipStream_t stream) {
    const float* q     = (const float*)d_in[0];
    const float* k     = (const float*)d_in[1];
    const float* v     = (const float*)d_in[2];
    const float* g_o   = (const float*)d_in[3];
    const float* g_lse = (const float*)d_in[4];
    const float* gq    = (const float*)d_in[5];
    const float* gk    = (const float*)d_in[6];
    const int*   biq   = (const int*)d_in[7];
    const int*   bikv  = (const int*)d_in[8];

    const int bsz = in_sizes[0] / (BQ * NUM_Q_HEAD * HEAD_DIM);
    const size_t o_elems   = (size_t)bsz * SEQLEN_Q * NUM_Q_HEAD * HEAD_DIM;
    const size_t lse_elems = (size_t)bsz * NUM_Q_HEAD * SEQLEN_Q;

    float* out_o   = (float*)d_out;
    float* out_lse = out_o + o_elems;

    // Bulk pass-through of unmodified state (bandwidth-bound: ~512MB @ 23.3TB/s).
    hipMemcpyAsync(out_o,   g_o,   o_elems   * sizeof(float), hipMemcpyDeviceToDevice, stream);
    hipMemcpyAsync(out_lse, g_lse, lse_elems * sizeof(float), hipMemcpyDeviceToDevice, stream);

    // One workgroup (8 wave32) per (batch, q-head) pair.
    dim3 grid(bsz * NUM_Q_HEAD);
    oswa_block_kernel<<<grid, 256, 0, stream>>>(q, k, v, g_o, g_lse, gq, gk,
                                                biq, bikv, out_o, out_lse);
}
